// HyperLSTM_35313221108066
// MI455X (gfx1250) — matrix operand order repaired and checked by reading, not compiled, not run on hardware
//
#include <hip/hip_runtime.h>
#include <math.h>

// ---------------------------------------------------------------------------
// HyperLSTM forward for MI455X (gfx1250, wave32, WMMA 16x16x32 bf16 + TDM).
//   - bf16 activation tiles pulled into LDS via TENSOR_LOAD_TO_LDS (TDM),
//     synchronized with s_wait_tensorcnt; fp32 weights converted to bf16
//     with vectorized float2 loads while staging into LDS (L2-resident).
//   - Input projections + decoder: LDS-staged bf16 WMMA GEMMs.
//   - Recurrence: 64 per-step kernels per layer, gate-fused WMMA tiles.
// ---------------------------------------------------------------------------

typedef __bf16 bf16_t;
typedef bf16_t v16bf __attribute__((ext_vector_type(16)));
typedef float  v8f   __attribute__((ext_vector_type(8)));
typedef unsigned short u16;
typedef unsigned int   u32;
typedef unsigned int u32x4 __attribute__((ext_vector_type(4)));
typedef int          i32x4 __attribute__((ext_vector_type(4)));
typedef int          i32x8 __attribute__((ext_vector_type(8)));

#define SCALE_STRIDE 4608   // row stride for ws_i / ws_h / bias arrays (max 4n)

__device__ __forceinline__ u16 f2bf(float x) {
  union { float f; u32 u; } a; a.f = x;
  u32 r = a.u + 0x7FFFu + ((a.u >> 16) & 1u);   // round-to-nearest-even
  return (u16)(r >> 16);
}

__device__ __forceinline__ u32 pack2bf(float lo, float hi) {
  return (u32)f2bf(lo) | ((u32)f2bf(hi) << 16);
}

__device__ __forceinline__ float sigf(float x) { return 1.0f / (1.0f + __expf(-x)); }

// ---------------------------------------------------------------------------
// TDM: async-load a 64-row x 32-element (2-byte elems) tile from global into
// LDS (rows packed 64B apart). D# per CDNA5 ISA:
//   group0: count=1, lds_addr, global_addr (tile start), type=2
//   group1: data_size=1 (2B), tensor dims == tile dims (no OOB), dim0 stride.
// Issued by one wave; completion via s_wait_tensorcnt 0 before the barrier.
// ---------------------------------------------------------------------------
__device__ __forceinline__ void tdm_load_tile_b16(u16* lds, const u16* gsrc,
                                                  u32 stride_elems) {
  const u32 TW = 32, TH = 64;
  u32 lds_off = (u32)(unsigned long long)lds;           // flat addr[31:0] == LDS offset
  unsigned long long ga = (unsigned long long)gsrc;
  u32x4 g0;
  g0.x = 1u;                                            // count=1, user mode
  g0.y = lds_off;                                       // lds_addr
  g0.z = (u32)ga;                                       // global_addr[31:0]
  g0.w = (u32)((ga >> 32) & 0x01ffffffu) | (2u << 30);  // addr[56:32], type=2
  i32x8 g1;
  g1[0] = (int)(1u << 16);                              // wg_mask=0, data_size=2B
  g1[1] = (int)((TW & 0xffffu) << 16);                  // tensor_dim0[15:0]
  g1[2] = (int)(((TW >> 16) & 0xffffu) | ((TH & 0xffffu) << 16));  // dim0 hi | dim1 lo
  g1[3] = (int)(((TH >> 16) & 0xffffu) | ((TW & 0xffffu) << 16));  // dim1 hi | tile_dim0
  g1[4] = (int)(TH & 0xffffu);                          // tile_dim1 (tile_dim2=0)
  g1[5] = (int)stride_elems;                            // tensor_dim0_stride[31:0]
  g1[6] = 0;
  g1[7] = 0;
  i32x4 z4 = {0, 0, 0, 0};
#if __has_include(<hip/amd_detail/amd_gfx1250_TDM.h>)
  i32x8 z8 = {0, 0, 0, 0, 0, 0, 0, 0};
  __builtin_amdgcn_tensor_load_to_lds(g0, g1, z4, z4, z8, 0);
#else
  __builtin_amdgcn_tensor_load_to_lds(g0, g1, z4, z4, 0);
#endif
}

// A fragment (16x32 bf16, M rows in LDS row-major, K contiguous, ld elements).
__device__ __forceinline__ v16bf frag_a(const u16* lds, int row0, int ld) {
  int lane = threadIdx.x & 31;
  int m = lane & 15, half = lane >> 4;
  union { v16bf v; u32 u[8]; } f;
  const u32* p = (const u32*)(lds + (row0 + m) * ld);
  int kb = half * 4;
#pragma unroll
  for (int i = 0; i < 4; ++i) { f.u[i] = p[kb + i]; f.u[4 + i] = p[8 + kb + i]; }
  return f.v;
}

// B fragment (32x16 bf16). LDS holds tile as [n][k] row-major (k contiguous).
__device__ __forceinline__ v16bf frag_b(const u16* lds, int col0, int ld) {
  int lane = threadIdx.x & 31;
  int n = lane & 15, half = lane >> 4;
  union { v16bf v; u32 u[8]; } f;
  const u32* p = (const u32*)(lds + (col0 + n) * ld);
  int kb = half * 4;
#pragma unroll
  for (int i = 0; i < 4; ++i) { f.u[i] = p[kb + i]; f.u[4 + i] = p[8 + kb + i]; }
  return f.v;
}

#define WMMA_BF16(acc, a, b) \
  acc = __builtin_amdgcn_wmma_f32_16x16x32_bf16(false, (a), false, (b), (short)0, (acc), false, false)

// ---------------------------------------------------------------------------
// Embedding: x[t,b,k] = elem_emb[tok,k] + (hnet[b] . emb_S[k]) * hnet_emb[tok,k]
// ---------------------------------------------------------------------------
__global__ void embed_kernel(const int* __restrict__ tokens,
                             const float* __restrict__ hnet,
                             const float* __restrict__ elem_emb,
                             const float* __restrict__ hnet_emb,
                             const float* __restrict__ emb_S,
                             u16* __restrict__ X0) {
  const int KP = 416;
  int idx = blockIdx.x * blockDim.x + threadIdx.x;
  if (idx >= 4096 * KP) return;
  int m = idx / KP, k = idx - m * KP;
  u16 out = 0;
  if (k < 400) {
    int b = m & 63;
    int tok = tokens[m];
    float sc = 0.f;
#pragma unroll
    for (int h = 0; h < 7; ++h) sc += hnet[b * 7 + h] * emb_S[k * 7 + h];
    out = f2bf(elem_emb[tok * 400 + k] + sc * hnet_emb[tok * 400 + k]);
  }
  X0[idx] = out;
}

// ---------------------------------------------------------------------------
// Hypernet scale vectors: ws_i, ws_h, and fused bias (per batch, per 4n col).
// ---------------------------------------------------------------------------
__global__ void scales_kernel(const float* __restrict__ hnet,
                              const float* __restrict__ S_i,
                              const float* __restrict__ S_h,
                              const float* __restrict__ be_i, const float* __restrict__ bh_i,
                              const float* __restrict__ be_h, const float* __restrict__ bh_h,
                              int n4,
                              float* __restrict__ ws_i, float* __restrict__ ws_h,
                              float* __restrict__ bias) {
  int idx = blockIdx.x * blockDim.x + threadIdx.x;
  if (idx >= 64 * n4) return;
  int b = idx / n4, c = idx - b * n4;
  float wi = 0.f, bi = 0.f, wh = 0.f, bh = 0.f;
#pragma unroll
  for (int h = 0; h < 7; ++h) {
    float hv = hnet[b * 7 + h];
    wi += hv * S_i[c * 7 + h];
    bi += hv * S_i[(n4 + c) * 7 + h];
    wh += hv * S_h[c * 7 + h];
    bh += hv * S_h[(n4 + c) * 7 + h];
  }
  ws_i[b * SCALE_STRIDE + c] = wi;
  ws_h[b * SCALE_STRIDE + c] = wh;
  bias[b * SCALE_STRIDE + c] = be_i[c] + bi * bh_i[c] + be_h[c] + bh * bh_h[c];
}

// ---------------------------------------------------------------------------
// h/c state init (h in bf16 with zero K-padding).
// ---------------------------------------------------------------------------
__global__ void init_state_kernel(const float* __restrict__ h0,
                                  const float* __restrict__ c0,
                                  int n, int npad,
                                  u16* __restrict__ Hbf, float* __restrict__ C) {
  int idx = blockIdx.x * blockDim.x + threadIdx.x;
  if (idx >= 64 * npad) return;
  int b = idx / npad, j = idx - b * npad;
  if (j < n) { Hbf[idx] = f2bf(h0[b * n + j]); C[idx] = c0[b * n + j]; }
  else       { Hbf[idx] = 0;                   C[idx] = 0.f; }
}

// ---------------------------------------------------------------------------
// Input projection for all timesteps (t-independent part of the LSTM):
//   PRE[m, j] = X[m]·We[j] + ws_i[b,j] * (X[m]·Wh[j]) + bias[b,j],  b = m%64
// M=4096, N=4n, K=din.  64x64 tile / workgroup, 8 waves, dual accumulators.
// A-tile via TDM; weight tiles via float2 loads + bf16 pack.
// ---------------------------------------------------------------------------
__global__ void __launch_bounds__(256)
input_proj_kernel(const u16* __restrict__ Xbf, int Kpad, int Kreal,
                  const float* __restrict__ We, const float* __restrict__ Wh,
                  int Nreal, int Nstride,
                  const float* __restrict__ ws_i, const float* __restrict__ bias,
                  float* __restrict__ PRE) {
  __shared__ u16 At[64 * 32];
  __shared__ u16 Bte[64 * 32];
  __shared__ u16 Bth[64 * 32];
  int tid = threadIdx.x;
  int wave = tid >> 5;
  int mtile = blockIdx.y * 64;
  int ntile = blockIdx.x * 64;
  int rb = (wave & 3) * 16;
  int cb = (wave >> 2) * 32;
  v8f acc_e0 = {}, acc_e1 = {}, acc_h0 = {}, acc_h1 = {};
  for (int k0 = 0; k0 < Kpad; k0 += 32) {
    if (tid < 32)   // wave 0 issues the async A-tile DMA
      tdm_load_tile_b16(At, Xbf + (size_t)mtile * Kpad + k0, (u32)Kpad);
#pragma unroll
    for (int it = 0; it < 4; ++it) {
      int p = tid + it * 256;       // dword slot: col(6b) x k-pair(4b)
      int c = p >> 4, kp = p & 15;
      int ng = ntile + c;
      int k = k0 + kp * 2;          // Kreal is even -> pair fully in/out
      u32 ve = 0, vh = 0;
      if (ng < Nreal && k < Kreal) {
        const float2 e2 = *(const float2*)(We + (size_t)ng * Kreal + k);
        const float2 h2 = *(const float2*)(Wh + (size_t)ng * Kreal + k);
        ve = pack2bf(e2.x, e2.y);
        vh = pack2bf(h2.x, h2.y);
      }
      ((u32*)Bte)[p] = ve;
      ((u32*)Bth)[p] = vh;
    }
    if (tid < 32) __builtin_amdgcn_s_wait_tensorcnt((short)0);
    __syncthreads();
    v16bf a  = frag_a(At, rb, 32);
    v16bf b0 = frag_b(Bte, cb, 32);      WMMA_BF16(acc_e0, a, b0);
    v16bf b1 = frag_b(Bte, cb + 16, 32); WMMA_BF16(acc_e1, a, b1);
    v16bf b2 = frag_b(Bth, cb, 32);      WMMA_BF16(acc_h0, a, b2);
    v16bf b3 = frag_b(Bth, cb + 16, 32); WMMA_BF16(acc_h1, a, b3);
    __syncthreads();
  }
  int lane = tid & 31, half = lane >> 4, nn = lane & 15;
#pragma unroll
  for (int r = 0; r < 8; ++r) {
    int m = mtile + rb + r + half * 8;
    int b = m & 63;
    int ng0 = ntile + cb + nn;
    int ng1 = ng0 + 16;
    if (ng0 < Nreal)
      PRE[m * Nstride + ng0] = acc_e0[r] + ws_i[b * SCALE_STRIDE + ng0] * acc_h0[r]
                             + bias[b * SCALE_STRIDE + ng0];
    if (ng1 < Nreal)
      PRE[m * Nstride + ng1] = acc_e1[r] + ws_i[b * SCALE_STRIDE + ng1] * acc_h1[r]
                             + bias[b * SCALE_STRIDE + ng1];
  }
}

// ---------------------------------------------------------------------------
// One LSTM timestep. Workgroup owns hidden cols [j0, j0+32); computes all 4
// gate columns with both We_h and ws_h-scaled Wh_h products (8 WMMA
// accumulators/wave), then fuses sigmoid/tanh + c/h update.
// H tile via TDM from the double-buffered bf16 state.
// ---------------------------------------------------------------------------
__global__ void __launch_bounds__(256)
lstm_step_kernel(int t, int n, int npad,
                 const float* __restrict__ PRE, int Nstride,
                 const float* __restrict__ WeH, const float* __restrict__ WhH,
                 const float* __restrict__ ws_h,
                 const u16* __restrict__ Hin, u16* __restrict__ Hout,
                 float* __restrict__ C, u16* __restrict__ Xnext) {
  __shared__ u16 Ht[64 * 32];
  __shared__ u16 Wt[8][32 * 32];   // [gate*2 + (0=We,1=Wh)][jj*32 + kk]
  int tid = threadIdx.x;
  int wave = tid >> 5;
  int wv_b = wave & 3;
  int wv_j = wave >> 2;
  int j0 = blockIdx.x * 32;
  v8f acc_e[4] = {};
  v8f acc_h[4] = {};
  for (int k0 = 0; k0 < npad; k0 += 32) {
    if (tid < 32)
      tdm_load_tile_b16(Ht, Hin + k0, (u32)npad);
#pragma unroll
    for (int it = 0; it < 16; ++it) {
      int p = tid + it * 256;       // 0..4095 dword slots across 8 tiles
      int tile = p >> 9;
      int rem = p & 511;            // jj(5b) x k-pair(4b)
      int jj = rem >> 4, kp = rem & 15;
      int g = tile >> 1, isH = tile & 1;
      int jg = j0 + jj;
      int k = k0 + kp * 2;          // n is even -> pair fully in/out
      u32 v = 0;
      if (jg < n && k < n) {
        const float* W = isH ? WhH : WeH;
        const float2 w2 = *(const float2*)(W + (size_t)(g * n + jg) * n + k);
        v = pack2bf(w2.x, w2.y);
      }
      ((u32*)Wt[tile])[rem] = v;
    }
    if (tid < 32) __builtin_amdgcn_s_wait_tensorcnt((short)0);
    __syncthreads();
    v16bf a = frag_a(Ht, wv_b * 16, 32);
#pragma unroll
    for (int g = 0; g < 4; ++g) {
      v16bf be = frag_b(Wt[g * 2 + 0], wv_j * 16, 32); WMMA_BF16(acc_e[g], a, be);
      v16bf bh = frag_b(Wt[g * 2 + 1], wv_j * 16, 32); WMMA_BF16(acc_h[g], a, bh);
    }
    __syncthreads();
  }
  int lane = tid & 31, half = lane >> 4, nn = lane & 15;
  int j = j0 + wv_j * 16 + nn;
#pragma unroll
  for (int r = 0; r < 8; ++r) {
    int b = wv_b * 16 + r + half * 8;
    int row = t * 64 + b;
    if (j < n) {
      float pre[4];
#pragma unroll
      for (int g = 0; g < 4; ++g) {
        int cc = g * n + j;
        pre[g] = PRE[row * Nstride + cc] + acc_e[g][r]
               + ws_h[b * SCALE_STRIDE + cc] * acc_h[g][r];
      }
      float fg = sigf(pre[0]);
      float ig = sigf(pre[1]);
      float og = sigf(pre[2]);
      float gg = tanhf(pre[3]);
      float cn = fg * C[b * npad + j] + ig * gg;
      C[b * npad + j] = cn;
      float hn = og * tanhf(cn);
      u16 hb = f2bf(hn);
      Hout[b * npad + j] = hb;
      Xnext[row * npad + j] = hb;
    } else if (j < npad) {
      Hout[b * npad + j] = 0;
      Xnext[row * npad + j] = 0;
    }
  }
}

// ---------------------------------------------------------------------------
// Decoder: logits[m, v] = X3[m]·elem_emb[v] + dec_bias[v]
// M=4096, N=33278, K=400 (padded 416). fp32 output to d_out.
// ---------------------------------------------------------------------------
__global__ void __launch_bounds__(256)
decoder_kernel(const u16* __restrict__ Xbf,
               const float* __restrict__ emb,
               const float* __restrict__ dec_bias,
               float* __restrict__ out) {
  __shared__ u16 At[64 * 32];
  __shared__ u16 Bt[64 * 32];
  const int KP = 416, KR = 400, NV = 33278;
  int tid = threadIdx.x;
  int wave = tid >> 5;
  int mtile = blockIdx.y * 64;
  int ntile = blockIdx.x * 64;
  int rb = (wave & 3) * 16;
  int cb = (wave >> 2) * 32;
  v8f acc0 = {}, acc1 = {};
  for (int k0 = 0; k0 < KP; k0 += 32) {
    if (tid < 32)
      tdm_load_tile_b16(At, Xbf + (size_t)mtile * KP + k0, (u32)KP);
#pragma unroll
    for (int it = 0; it < 4; ++it) {
      int p = tid + it * 256;
      int c = p >> 4, kp = p & 15;
      int vg = ntile + c;
      int k = k0 + kp * 2;          // KR=400 even
      u32 v = 0;
      if (vg < NV && k < KR) {
        const float2 w2 = *(const float2*)(emb + (size_t)vg * KR + k);
        v = pack2bf(w2.x, w2.y);
      }
      ((u32*)Bt)[p] = v;
    }
    if (tid < 32) __builtin_amdgcn_s_wait_tensorcnt((short)0);
    __syncthreads();
    v16bf a  = frag_a(At, rb, 32);
    v16bf b0 = frag_b(Bt, cb, 32);      WMMA_BF16(acc0, a, b0);
    v16bf b1 = frag_b(Bt, cb + 16, 32); WMMA_BF16(acc1, a, b1);
    __syncthreads();
  }
  int lane = tid & 31, half = lane >> 4, nn = lane & 15;
#pragma unroll
  for (int r = 0; r < 8; ++r) {
    size_t m = (size_t)(mtile + rb + r + half * 8);
    int v0 = ntile + cb + nn, v1 = v0 + 16;
    if (v0 < NV) out[m * NV + v0] = acc0[r] + dec_bias[v0];
    if (v1 < NV) out[m * NV + v1] = acc1[r] + dec_bias[v1];
  }
}

// ---------------------------------------------------------------------------
extern "C" void kernel_launch(void* const* d_in, const int* in_sizes, int n_in,
                              void* d_out, int out_size, void* d_ws, size_t ws_size,
                              hipStream_t stream) {
  (void)in_sizes; (void)n_in; (void)out_size; (void)ws_size;

  const int*   tokens   = (const int*)d_in[0];
  const float* hnet     = (const float*)d_in[1];
  const float* elem_emb = (const float*)d_in[2];
  const float* hnet_emb = (const float*)d_in[3];
  const float* emb_S    = (const float*)d_in[4];
  const float* dec_bias = (const float*)d_in[41];

  const int din_[3]  = {400, 1150, 1150};
  const int dinp_[3] = {416, 1152, 1152};
  const int n_[3]    = {1150, 1150, 400};
  const int np_[3]   = {1152, 1152, 416};
  const int n4_[3]   = {4600, 4600, 1600};
  const int n4p_[3]  = {4608, 4608, 1600};

  // Workspace carve-out (~112 MB).
  char* w = (char*)d_ws;
  auto carve = [&](size_t bytes) { void* p = (void*)w; w += (bytes + 255) & ~(size_t)255; return p; };
  u16*   X0   = (u16*)carve((size_t)4096 * 416 * 2);
  u16*   X1   = (u16*)carve((size_t)4096 * 1152 * 2);
  u16*   X2   = (u16*)carve((size_t)4096 * 1152 * 2);
  u16*   X3   = (u16*)carve((size_t)4096 * 416 * 2);
  float* PRE  = (float*)carve((size_t)4096 * 4608 * 4);
  float *wsi[3], *wsh[3], *bias[3];
  for (int l = 0; l < 3; ++l) {
    wsi[l]  = (float*)carve((size_t)64 * SCALE_STRIDE * 4);
    wsh[l]  = (float*)carve((size_t)64 * SCALE_STRIDE * 4);
    bias[l] = (float*)carve((size_t)64 * SCALE_STRIDE * 4);
  }
  u16*   Hb0 = (u16*)carve((size_t)64 * 1152 * 2);
  u16*   Hb1 = (u16*)carve((size_t)64 * 1152 * 2);
  float* Cst = (float*)carve((size_t)64 * 1152 * 4);

  u16* Xin_[3]  = {X0, X1, X2};
  u16* Xout_[3] = {X1, X2, X3};

  // 1) Embeddings -> X0 (bf16, padded).
  embed_kernel<<<(4096 * 416 + 255) / 256, 256, 0, stream>>>(
      tokens, hnet, elem_emb, hnet_emb, emb_S, X0);

  // 2) Hypernet scale vectors per layer.
  for (int l = 0; l < 3; ++l) {
    int base = 5 + 12 * l;
    scales_kernel<<<(64 * n4_[l] + 255) / 256, 256, 0, stream>>>(
        hnet,
        (const float*)d_in[base + 4],   // i2h_S
        (const float*)d_in[base + 9],   // h2h_S
        (const float*)d_in[base + 2], (const float*)d_in[base + 3],   // be_i, bh_i
        (const float*)d_in[base + 7], (const float*)d_in[base + 8],   // be_h, bh_h
        n4_[l], wsi[l], wsh[l], bias[l]);
  }

  // 3) Layers: input projection for all t, then sequential scan.
  for (int l = 0; l < 3; ++l) {
    int base = 5 + 12 * l;
    const float* We_i = (const float*)d_in[base + 0];
    const float* Wh_i = (const float*)d_in[base + 1];
    const float* We_h = (const float*)d_in[base + 5];
    const float* Wh_h = (const float*)d_in[base + 6];
    const float* h0   = (const float*)d_in[base + 10];
    const float* c0   = (const float*)d_in[base + 11];

    init_state_kernel<<<(64 * np_[l] + 255) / 256, 256, 0, stream>>>(
        h0, c0, n_[l], np_[l], Hb0, Cst);

    dim3 g1(n4p_[l] / 64, 64);
    input_proj_kernel<<<g1, 256, 0, stream>>>(
        Xin_[l], dinp_[l], din_[l], We_i, Wh_i,
        n4_[l], n4p_[l], wsi[l], bias[l], PRE);

    for (int t = 0; t < 64; ++t) {
      u16* hin  = (t & 1) ? Hb1 : Hb0;
      u16* hout = (t & 1) ? Hb0 : Hb1;
      lstm_step_kernel<<<np_[l] / 32, 256, 0, stream>>>(
          t, n_[l], np_[l], PRE, n4p_[l], We_h, Wh_h, wsh[l],
          hin, hout, Cst, Xout_[l]);
    }
  }

  // 4) Decoder GEMM -> d_out (4096 x 33278 fp32).
  dim3 g3((33278 + 63) / 64, 64);
  decoder_kernel<<<g3, 256, 0, stream>>>(X3, elem_emb, dec_bias, (float*)d_out);
}
